// PairBoundaryPredictor_84610855731721
// MI455X (gfx1250) — matrix operand order repaired and checked by reading, not compiled, not run on hardware
//
#include <hip/hip_runtime.h>
#include <stdint.h>

#define B_ 2
#define F_ 128
#define KN_ 12
#define P_ 128
#define D_ 256
#define L_ 6
#define H_ 8
#define DFF_ 1024
#define T_ 64
#define CIN_ 42
#define CH_ 64
#define NT 256       // B*F tokens
#define NP 256       // B*P pairs/images
#define NPIX 4096    // T*T

typedef __bf16 bf16t;
typedef bf16t v16bf __attribute__((ext_vector_type(16)));
typedef bf16t v8bf  __attribute__((ext_vector_type(8)));
typedef float v8f   __attribute__((ext_vector_type(8)));

// ---------- scalar helpers ----------
__device__ __forceinline__ unsigned short f2bf(float f) {
    unsigned int u = __float_as_uint(f);
    unsigned int r = u + 0x7fffu + ((u >> 16) & 1u);
    return (unsigned short)(r >> 16);
}
__device__ __forceinline__ bf16t us2bf(unsigned short u) {
    union { unsigned short s; bf16t b; } x; x.s = u; return x.b;
}
__device__ __forceinline__ v16bf cat8(v8bf lo, v8bf hi) {
    return __builtin_shufflevector(lo, hi, 0, 1, 2, 3, 4, 5, 6, 7,
                                           8, 9, 10, 11, 12, 13, 14, 15);
}
__device__ __forceinline__ float gelu_f(float x) {
    return 0.5f * x * (1.f + erff(x * 0.7071067811865475f));
}
__device__ __forceinline__ float silu_f(float x) {
    return x / (1.f + expf(-x));
}
__device__ __forceinline__ float sigm_f(float x) {
    return 1.f / (1.f + expf(-x));
}
__device__ __forceinline__ float blk_reduce(float v, float* buf) {
    int t = threadIdx.x;
    buf[t] = v; __syncthreads();
    for (int s = 128; s > 0; s >>= 1) { if (t < s) buf[t] += buf[t + s]; __syncthreads(); }
    float r = buf[0]; __syncthreads();
    return r;
}
// A-matrix fragment permutation: element (row,k) of an [M][K] bf16 matrix
// stored so that each WMMA lane's 16 halfs are one contiguous 32B run.
// layout: [(row>>4)][(k>>5)] tiles of 512 halfs; within tile: lane*16 + half.
// lane = (row&15) + 16*g with g=(kk>>3)&1; half j=(kk&7)+((kk&16)?8:0), kk=k&31.
__device__ __forceinline__ size_t afrag_idx(int row, int k, int K) {
    int kk = k & 31;
    int g = (kk >> 3) & 1;
    int j = (kk & 7) + ((kk & 16) ? 8 : 0);
    return (((size_t)(row >> 4) * (K >> 5) + (k >> 5)) << 9) +
           (((row & 15) + (g << 4)) << 4) + j;
}

// ---------- weight packing into WMMA B-fragment order ----------
// layout: out[ ((k>>5)*N + n)*32 + 16*((k>>3)&1) + (k&7) + 8*((k>>4)&1) ]  (bf16)
__global__ void __launch_bounds__(256) pack_wT_frag(const float* __restrict__ W,
                                                    unsigned short* __restrict__ out,
                                                    int O, int I) { // W[O][I], O=N, I=K
    int i = blockIdx.x * 256 + threadIdx.x;
    if (i >= O * I) return;
    int n = i / I, k = i % I;
    out[(((size_t)(k >> 5) * O + n) << 5) | (((k >> 3) & 1) << 4) | (k & 7) | ((k >> 1) & 8)] = f2bf(W[i]);
}
// conv W[co][ci][ky][kx], k=(ky*KS+kx)*Cin+ci, zero padded to Kpad, fragment order
__global__ void __launch_bounds__(256) pack_conv_frag(const float* __restrict__ W,
                                                      unsigned short* __restrict__ out,
                                                      int Cin, int KS, int Kpad) {
    int i = blockIdx.x * 256 + threadIdx.x;
    if (i >= Kpad * 64) return;
    int k = i >> 6, co = i & 63;
    unsigned short v = 0;
    int KK = KS * KS * Cin;
    if (k < KK) {
        int ci = k % Cin; int t2 = k / Cin; int kx = t2 % KS; int ky = t2 / KS;
        v = f2bf(W[((size_t)(co * Cin + ci) * KS + ky) * KS + kx]);
    }
    out[(((size_t)(k >> 5) * 64 + co) << 5) | (((k >> 3) & 1) << 4) | (k & 7) | ((k >> 1) & 8)] = v;
}

// ---------- bf16 WMMA GEMM, fully fragment-packed operands ----------
// Ap: A in afrag layout [(M/16)][(K/32)][512]; Bp: [(K/32)][N][32].
// No LDS, no barriers: inner loop = coalesced b128 loads + WMMA.
// Each wave: 2 M-tiles sharing one B fragment. Block = 64(M) x 64(N).
__global__ void __launch_bounds__(256) gemm_bf16(
    const unsigned short* __restrict__ Ap,
    const unsigned short* __restrict__ Bp, int N, int K,
    const float* __restrict__ bias, const float* __restrict__ resid,
    float* __restrict__ outf, unsigned short* __restrict__ outbf,
    int ldc, int act) {
    const int tid = threadIdx.x;
    const int wave = tid >> 5, lane = tid & 31;
    const int n0 = blockIdx.y * 64;
    const int nt = (wave & 3) * 16;
    const int gn = n0 + nt + (lane & 15);
    const int kf = (lane >> 4) << 4;
    const int KB = K >> 5;
    const int mtg0 = (blockIdx.x << 2) + ((wave >> 2) << 1); // global m-tile (pair)
    v8f acc[2];
    acc[0] = (v8f){0.f, 0.f, 0.f, 0.f, 0.f, 0.f, 0.f, 0.f};
    acc[1] = acc[0];
    const unsigned short* a0p = Ap + (((size_t)mtg0 * KB) << 9) + (lane << 4);
    const unsigned short* a1p = a0p + ((size_t)KB << 9);
    for (int kb = 0; kb < KB; ++kb) {
        v16bf a0 = *(const v16bf*)(a0p + ((size_t)kb << 9));
        v16bf a1 = *(const v16bf*)(a1p + ((size_t)kb << 9));
        v16bf b  = *(const v16bf*)&Bp[(((size_t)kb * N + gn) << 5) + kf];
        acc[0] = __builtin_amdgcn_wmma_f32_16x16x32_bf16(false, a0, false, b, (short)0, acc[0], false, false);
        acc[1] = __builtin_amdgcn_wmma_f32_16x16x32_bf16(false, a1, false, b, (short)0, acc[1], false, false);
    }
    float bv = bias ? bias[gn] : 0.f;
    #pragma unroll
    for (int t = 0; t < 2; ++t) {
        int mbase = (mtg0 + t) * 16 + ((lane >> 4) << 3);
        #pragma unroll
        for (int r = 0; r < 8; ++r) {
            int gm = mbase + r;
            float v = acc[t][r] + bv;
            if (act == 1) v = gelu_f(v);
            if (resid) v += resid[(size_t)gm * ldc + gn];
            if (outf)  outf[(size_t)gm * ldc + gn] = v;
            if (outbf) outbf[afrag_idx(gm, gn, ldc)] = f2bf(v);
        }
    }
}

// ---------- implicit-GEMM conv (KSxKS, pad), 64 out channels, WMMA ----------
template <int KS, int PAD, int CIN>
__global__ void __launch_bounds__(256) conv_wmma(
    const unsigned short* __restrict__ X,   // [img][CIN][64][64] bf16
    const unsigned short* __restrict__ Wp,  // fragment-packed [KPAD/32][64][32] bf16
    const float* __restrict__ bias,         // [64]
    float* __restrict__ Y) {                // [img][64][4096] fp32
    constexpr int W2 = (KS == 3) ? 34 : 32;
    constexpr int KK = KS * KS * CIN;
    constexpr int KPAD = (KK + 31) & ~31;
    __shared__ __align__(16) unsigned short sIn[3 * 34 * 64];
    const int tid = threadIdx.x;
    const int blk = blockIdx.x;
    const int img = blk >> 7;             // 128 tiles per image
    const int tile = blk & 127;
    const int oy = tile >> 1;
    const int ox0 = (tile & 1) * 32;
    // stage input halo tile [KS rows][W2 cols][CIN]  (once per block)
    for (int e = tid; e < KS * W2 * CIN; e += 256) {
        int ci = e % CIN; int rc = e / CIN; int col = rc % W2; int row = rc / W2;
        int y = oy - PAD + row, x = ox0 - PAD + col;
        unsigned short v = 0;
        if (y >= 0 && y < T_ && x >= 0 && x < T_)
            v = X[((size_t)img * CIN + ci) * NPIX + y * T_ + x];
        sIn[(row * W2 + col) * CIN + ci] = v;
    }
    __syncthreads();
    const int wave = tid >> 5, lane = tid & 31;
    const int mt = (wave >> 2) * 16, nt = (wave & 3) * 16;
    const int m = mt + (lane & 15);
    const int kb = (lane >> 4) << 3;
    const int n = nt + (lane & 15);
    const int kf = (lane >> 4) << 4;
    v8f acc = {0.f, 0.f, 0.f, 0.f, 0.f, 0.f, 0.f, 0.f};
    for (int k0 = 0; k0 < KPAD; k0 += 32) {
        v16bf a;
        if constexpr ((CIN & 7) == 0) {
            // runs of 8 never cross a (ky,kx) cell: wide ds loads
            v8bf parts[2];
            #pragma unroll
            for (int h = 0; h < 2; ++h) {
                int k = k0 + kb + h * 16;
                int ci = k % CIN;
                int cell = k / CIN;
                int kx = cell % KS, ky = cell / KS;
                parts[h] = *(const v8bf*)&sIn[(ky * W2 + m + kx) * CIN + ci];
            }
            a = cat8(parts[0], parts[1]);
        } else {
            #pragma unroll
            for (int j = 0; j < 16; ++j) {
                int k = k0 + (j < 8 ? j : j + 8) + kb;
                unsigned short u = 0;
                if (k < KK) {
                    int ci = k % CIN; int t2 = k / CIN; int kx = t2 % KS; int ky = t2 / KS;
                    u = sIn[(ky * W2 + (m + kx)) * CIN + ci];
                }
                a[j] = us2bf(u);
            }
        }
        v16bf b = *(const v16bf*)&Wp[(((size_t)(k0 >> 5) * 64 + n) << 5) + kf];
        acc = __builtin_amdgcn_wmma_f32_16x16x32_bf16(false, a, false, b, (short)0, acc, false, false);
    }
    float bv = bias[n];
    int mb = mt + ((lane >> 4) << 3);
    #pragma unroll
    for (int r = 0; r < 8; ++r) {
        int pix = oy * T_ + ox0 + mb + r;
        Y[((size_t)img * CH_ + n) * NPIX + pix] = acc[r] + bv;
    }
}

// ---------- node embedding + continuous MLP ----------
__global__ void __launch_bounds__(256) node_embed(
    const int* __restrict__ nf, const float* __restrict__ cent, const float* __restrict__ bbox,
    const float* __restrict__ e0, const float* __restrict__ e1, const float* __restrict__ e2,
    const float* __restrict__ cw1, const float* __restrict__ cb1,
    const float* __restrict__ cw2, const float* __restrict__ cb2,
    float* __restrict__ nh) {
    int t = blockIdx.x, d = threadIdx.x;
    int i0 = nf[t * 3 + 0], i1 = nf[t * 3 + 1], i2 = nf[t * 3 + 2];
    float v = e0[i0 * D_ + d] + e1[i1 * D_ + d] + e2[i2 * D_ + d];
    float c[6] = {cent[t * 2], cent[t * 2 + 1], bbox[t * 4], bbox[t * 4 + 1], bbox[t * 4 + 2], bbox[t * 4 + 3]};
    float hs = cb1[d];
    #pragma unroll
    for (int j = 0; j < 6; ++j) hs += c[j] * cw1[d * 6 + j];
    __shared__ float sh[D_];
    sh[d] = gelu_f(hs); __syncthreads();
    float o = cb2[d];
    for (int j = 0; j < D_; ++j) o += sh[j] * cw2[d * D_ + j];
    nh[(size_t)t * D_ + d] = v + o;
}

// ---------- neighbor messages + positional ----------
__global__ void __launch_bounds__(256) message_k(
    const float* __restrict__ nh, const int* __restrict__ nidx, const int* __restrict__ ntok,
    const float* __restrict__ nmask, const float* __restrict__ edge, const float* __restrict__ pos,
    float* __restrict__ hid) {
    int t = blockIdx.x; int b = t >> 7; int f = t & 127;
    int d = threadIdx.x;
    float msum = 0.f, mcnt = 0.f;
    #pragma unroll
    for (int k = 0; k < KN_; ++k) {
        float mk = nmask[t * KN_ + k];
        mcnt += mk;
        int gi = nidx[t * KN_ + k]; if (gi < 0) gi = 0;
        int tk = ntok[t * KN_ + k];
        msum += mk * (nh[(size_t)(b * F_ + gi) * D_ + d] + edge[(size_t)tk * D_ + d]);
    }
    float denom = fmaxf(mcnt, 1.f);
    hid[(size_t)t * D_ + d] = nh[(size_t)t * D_ + d] + msum / denom + pos[f * D_ + d];
}

// ---------- layernorm (row of 256); bf16 out in A-fragment order (K=256) ----------
__global__ void __launch_bounds__(256) ln_kernel(
    const float* __restrict__ X, const float* __restrict__ s, const float* __restrict__ b,
    float* __restrict__ outf, unsigned short* __restrict__ outbf) {
    __shared__ float red[256];
    int row = blockIdx.x, d = threadIdx.x;
    float v = X[(size_t)row * D_ + d];
    float mean = blk_reduce(v, red) * (1.f / D_);
    float c = v - mean;
    float var = blk_reduce(c * c, red) * (1.f / D_);
    float r = rsqrtf(var + 1e-5f);
    float y = c * r * s[d] + b[d];
    if (outf)  outf[(size_t)row * D_ + d] = y;
    if (outbf) outbf[afrag_idx(row, d, D_)] = f2bf(y);
}

// ---------- attention (one block per (b,h)); bf16 out in A-fragment order ----------
__global__ void __launch_bounds__(128) attn_kernel(
    const float* __restrict__ qkv, const unsigned char* __restrict__ fmask,
    unsigned short* __restrict__ obf) {
    __shared__ float sK[F_ * 32];
    __shared__ float sV[F_ * 32];
    __shared__ float sM[F_];
    int bh = blockIdx.x; int b = bh >> 3; int h = bh & 7;
    int f = threadIdx.x;
    int row = b * F_ + f;
    const float* qp = qkv + (size_t)row * 768 + h * 32;
    float qr[32];
    #pragma unroll
    for (int d = 0; d < 32; ++d) {
        qr[d] = qp[d];
        sK[f * 32 + d] = qp[256 + d];
        sV[f * 32 + d] = qp[512 + d];
    }
    sM[f] = fmask[b * F_ + f] ? 0.f : -1e9f;
    __syncthreads();
    const float scale = 0.17677669529663687f; // 1/sqrt(32)
    float mx = -3.4e38f;
    for (int j = 0; j < F_; ++j) {
        float s = 0.f;
        #pragma unroll
        for (int d = 0; d < 32; ++d) s += qr[d] * sK[j * 32 + d];
        s = s * scale + sM[j];
        mx = fmaxf(mx, s);
    }
    float den = 0.f, accv[32];
    #pragma unroll
    for (int d = 0; d < 32; ++d) accv[d] = 0.f;
    for (int j = 0; j < F_; ++j) {
        float s = 0.f;
        #pragma unroll
        for (int d = 0; d < 32; ++d) s += qr[d] * sK[j * 32 + d];
        s = s * scale + sM[j];
        float e = expf(s - mx);
        den += e;
        #pragma unroll
        for (int d = 0; d < 32; ++d) accv[d] += e * sV[j * 32 + d];
    }
    float inv = 1.f / den;
    #pragma unroll
    for (int d = 0; d < 32; ++d)
        obf[afrag_idx(row, h * 32 + d, D_)] = f2bf(accv[d] * inv);
}

// ---------- pair feature MLP ----------
__global__ void __launch_bounds__(256) pair_pf_kernel(
    const float* __restrict__ pfeat, const float* __restrict__ w1, const float* __restrict__ b1,
    const float* __restrict__ w2, const float* __restrict__ b2, float* __restrict__ out) {
    __shared__ float sh[D_];
    __shared__ float sf[13];
    int p = blockIdx.x, d = threadIdx.x;
    if (d < 13) sf[d] = pfeat[p * 13 + d];
    __syncthreads();
    float s = b1[d];
    #pragma unroll
    for (int j = 0; j < 13; ++j) s += sf[j] * w1[d * 13 + j];
    sh[d] = gelu_f(s); __syncthreads();
    float o = b2[d];
    for (int j = 0; j < D_; ++j) o += sh[j] * w2[d * D_ + j];
    out[(size_t)p * D_ + d] = o;
}

// ---------- pair MLP -> token(32) ----------
__global__ void __launch_bounds__(256) pair_mlp_kernel(
    const float* __restrict__ hid, const float* __restrict__ pf, const int* __restrict__ pidx,
    const float* __restrict__ bh, const float* __restrict__ w1, const float* __restrict__ b1,
    const float* __restrict__ w2, const float* __restrict__ b2,
    const float* __restrict__ tw, const float* __restrict__ tb, float* __restrict__ token) {
    __shared__ float ph[1024];
    __shared__ float h1[128];
    __shared__ float h2[128];
    int p = blockIdx.x; int b = p >> 7;
    int d = threadIdx.x;
    int iu = pidx[p * 2 + 0]; int iv = pidx[p * 2 + 1];
    bool border = iv < 0;
    if (iu < 0) iu = 0; if (iv < 0) iv = 0;
    float uh = hid[(size_t)(b * F_ + iu) * D_ + d];
    float vh = border ? bh[d] : hid[(size_t)(b * F_ + iv) * D_ + d];
    ph[d] = uh; ph[256 + d] = vh; ph[512 + d] = fabsf(uh - vh); ph[768 + d] = pf[(size_t)p * D_ + d];
    __syncthreads();
    if (d < 128) {
        float s = b1[d];
        for (int j = 0; j < 1024; ++j) s += ph[j] * w1[d * 1024 + j];
        h1[d] = gelu_f(s);
    }
    __syncthreads();
    if (d < 128) {
        float s = b2[d];
        for (int j = 0; j < 128; ++j) s += h1[j] * w2[d * 128 + j];
        h2[d] = gelu_f(s);
    }
    __syncthreads();
    if (d < 32) {
        float s = tb[d];
        for (int j = 0; j < 128; ++j) s += h2[j] * tw[d * 128 + j];
        token[p * 32 + d] = s;
    }
}

// ---------- rasterize 42-channel input maps (bf16) ----------
__global__ void __launch_bounds__(256) build_maps_kernel(
    const float* __restrict__ token, const int* __restrict__ pidx, const float* __restrict__ pfeat,
    const float* __restrict__ cent, const float* __restrict__ bbox, unsigned short* __restrict__ Xout) {
    int bc = blockIdx.x;
    int img = bc / CIN_; int c = bc % CIN_;
    int b = img >> 7;
    int iu = pidx[img * 2 + 0]; int iv = pidx[img * 2 + 1];
    bool border = iv < 0;
    if (iu < 0) iu = 0; if (iv < 0) iv = 0;
    float nb = border ? 0.f : 1.f;
    const float tau = 1.5f / 64.f, sig = 1.25f / 64.f;
    float ucx = cent[(b * F_ + iu) * 2 + 0], ucy = cent[(b * F_ + iu) * 2 + 1];
    float vcx = border ? ucx : cent[(b * F_ + iv) * 2 + 0];
    float vcy = border ? ucy : cent[(b * F_ + iv) * 2 + 1];
    float ub[4], vb[4];
    #pragma unroll
    for (int i = 0; i < 4; ++i) {
        ub[i] = bbox[(b * F_ + iu) * 4 + i];
        vb[i] = border ? ub[i] : bbox[(b * F_ + iv) * 4 + i];
    }
    const float* pfr = pfeat + (size_t)img * 13;
    float tokv = (c < 32) ? token[img * 32 + c] : 0.f;
    for (int i = threadIdx.x; i < NPIX; i += 256) {
        int iy = i >> 6, ix = i & 63;
        float gx = (ix + 0.5f) / 64.f, gy = (iy + 0.5f) / 64.f;
        float val;
        if (c < 32) {
            val = tokv;
        } else if (c == 32) {
            val = sigm_f((gx - ub[0]) / tau) * sigm_f((ub[2] - gx) / tau) *
                  sigm_f((gy - ub[1]) / tau) * sigm_f((ub[3] - gy) / tau);
        } else if (c == 33) {
            val = sigm_f((gx - vb[0]) / tau) * sigm_f((vb[2] - gx) / tau) *
                  sigm_f((gy - vb[1]) / tau) * sigm_f((vb[3] - gy) / tau) * nb;
        } else if (c == 34) {
            val = sigm_f((gx - pfr[4]) / tau) * sigm_f((pfr[6] - gx) / tau) *
                  sigm_f((gy - pfr[5]) / tau) * sigm_f((pfr[7] - gy) / tau);
        } else if (c == 35) {
            float dx = (gx - ucx) / sig, dy = (gy - ucy) / sig;
            val = expf(-0.5f * (dx * dx + dy * dy));
        } else if (c == 36) {
            float dx = (gx - vcx) / sig, dy = (gy - vcy) / sig;
            val = expf(-0.5f * (dx * dx + dy * dy)) * nb;
        } else {
            val = pfr[8 + (c - 37)];
        }
        Xout[((size_t)img * CIN_ + c) * NPIX + i] = f2bf(val);
    }
}

// ---------- groupnorm stats (one block per (img,group)) ----------
__global__ void __launch_bounds__(256) gn_stats(const float* __restrict__ X,
                                                float* __restrict__ gm, float* __restrict__ gr) {
    __shared__ float red[256];
    int g = blockIdx.x;
    const float* p = X + (size_t)g * 32768;
    float s = 0.f, ss = 0.f;
    for (int i = threadIdx.x; i < 32768; i += 256) { float v = p[i]; s += v; ss += v * v; }
    float ts = blk_reduce(s, red);
    float tss = blk_reduce(ss, red);
    if (threadIdx.x == 0) {
        float mean = ts * (1.f / 32768.f);
        float var = tss * (1.f / 32768.f) - mean * mean;
        gm[g] = mean;
        gr[g] = rsqrtf(var + 1e-5f);
    }
}

// ---------- groupnorm apply + SiLU (+resid), fp32/bf16 outs ----------
__global__ void __launch_bounds__(256) gn_apply(
    const float* __restrict__ X, const float* __restrict__ gm, const float* __restrict__ gr,
    const float* __restrict__ s, const float* __restrict__ b, const float* __restrict__ resid,
    float* __restrict__ outf, unsigned short* __restrict__ outbf) {
    int ic = blockIdx.x; int img = ic >> 6; int c = ic & 63;
    int g = img * 8 + (c >> 3);
    float mean = gm[g], r = gr[g], sc = s[c], bc = b[c];
    size_t base = (size_t)ic * NPIX;
    for (int i = threadIdx.x; i < NPIX; i += 256) {
        float v = (X[base + i] - mean) * r * sc + bc;
        v = silu_f(v);
        if (resid) v += resid[base + i];
        if (outf)  outf[base + i] = v;
        if (outbf) outbf[base + i] = f2bf(v);
    }
}

// ---------- 1x1 output conv (64->1) * pair_valid ----------
__global__ void __launch_bounds__(256) out_conv(
    const float* __restrict__ X, const float* __restrict__ w, const float* __restrict__ bias,
    const float* __restrict__ pvalid, float* __restrict__ out) {
    int i = blockIdx.x * 256 + threadIdx.x;
    if (i >= NP * NPIX) return;
    int img = i >> 12; int p = i & 4095;
    float s = bias[0];
    #pragma unroll 8
    for (int c = 0; c < CH_; ++c) s += X[((size_t)img * CH_ + c) * NPIX + p] * w[c];
    out[i] = s * pvalid[img];
}

// ==================== host launch ====================
extern "C" void kernel_launch(void* const* d_in, const int* in_sizes, int n_in,
                              void* d_out, int out_size, void* d_ws, size_t ws_size,
                              hipStream_t stream) {
    (void)in_sizes; (void)n_in; (void)out_size; (void)ws_size;
    const int*   node_features = (const int*)d_in[0];
    const unsigned char* face_mask = (const unsigned char*)d_in[1];
    const int*   nidx   = (const int*)d_in[2];
    const int*   ntok   = (const int*)d_in[3];
    const float* nmask  = (const float*)d_in[4];
    const float* cent   = (const float*)d_in[5];
    const float* bbox   = (const float*)d_in[6];
    const int*   pidx   = (const int*)d_in[7];
    const float* pfeat  = (const float*)d_in[8];
    const float* pvalid = (const float*)d_in[9];
    const float* emb0   = (const float*)d_in[10];
    const float* emb1   = (const float*)d_in[11];
    const float* emb2   = (const float*)d_in[12];
    const float* edge_emb = (const float*)d_in[13];
    const float* pos_emb  = (const float*)d_in[14];
    const float* border_hidden = (const float*)d_in[15];
    const float* cont_w1 = (const float*)d_in[16];
    const float* cont_b1 = (const float*)d_in[17];
    const float* cont_w2 = (const float*)d_in[18];
    const float* cont_b2 = (const float*)d_in[19];
    const float* enc_wqkv = (const float*)d_in[20];
    const float* enc_bqkv = (const float*)d_in[21];
    const float* enc_wo   = (const float*)d_in[22];
    const float* enc_bo   = (const float*)d_in[23];
    const float* ln1s = (const float*)d_in[24];
    const float* ln1b = (const float*)d_in[25];
    const float* ln2s = (const float*)d_in[26];
    const float* ln2b = (const float*)d_in[27];
    const float* fw1  = (const float*)d_in[28];
    const float* fb1  = (const float*)d_in[29];
    const float* fw2  = (const float*)d_in[30];
    const float* fb2  = (const float*)d_in[31];
    const float* fn_s = (const float*)d_in[32];
    const float* fn_b = (const float*)d_in[33];
    const float* pf_w1 = (const float*)d_in[34];
    const float* pf_b1 = (const float*)d_in[35];
    const float* pf_w2 = (const float*)d_in[36];
    const float* pf_b2 = (const float*)d_in[37];
    const float* pm_w1 = (const float*)d_in[38];
    const float* pm_b1 = (const float*)d_in[39];
    const float* pm_w2 = (const float*)d_in[40];
    const float* pm_b2 = (const float*)d_in[41];
    const float* tp_w  = (const float*)d_in[42];
    const float* tp_b  = (const float*)d_in[43];
    const float* c1w = (const float*)d_in[44]; const float* c1b = (const float*)d_in[45];
    const float* g1s = (const float*)d_in[46]; const float* g1b = (const float*)d_in[47];
    const float* c2w = (const float*)d_in[48]; const float* c2b = (const float*)d_in[49];
    const float* g2s = (const float*)d_in[50]; const float* g2b = (const float*)d_in[51];
    const float* skw = (const float*)d_in[52]; const float* skb = (const float*)d_in[53];
    const float* m1w = (const float*)d_in[54]; const float* m1b = (const float*)d_in[55];
    const float* mg1s = (const float*)d_in[56]; const float* mg1b = (const float*)d_in[57];
    const float* m2w = (const float*)d_in[58]; const float* m2b = (const float*)d_in[59];
    const float* mg2s = (const float*)d_in[60]; const float* mg2b = (const float*)d_in[61];
    const float* ow = (const float*)d_in[62]; const float* ob = (const float*)d_in[63];

    // deterministic workspace carve
    char* cur = (char*)d_ws;
    auto alloc = [&](size_t bytes) -> char* {
        char* p = cur; cur += (bytes + 255) & ~(size_t)255; return p;
    };
    float* nh   = (float*)alloc((size_t)NT * D_ * 4);
    float* hid  = (float*)alloc((size_t)NT * D_ * 4);
    float* hidf = (float*)alloc((size_t)NT * D_ * 4);
    unsigned short* tAbf = (unsigned short*)alloc((size_t)NT * 1024 * 2);
    unsigned short* tBbf = (unsigned short*)alloc((size_t)NT * 1024 * 2);
    float* tF1 = (float*)alloc((size_t)NT * 1024 * 4);
    unsigned short* qkvp = (unsigned short*)alloc((size_t)L_ * D_ * 768 * 2);
    unsigned short* wop  = (unsigned short*)alloc((size_t)L_ * D_ * D_ * 2);
    unsigned short* f1p  = (unsigned short*)alloc((size_t)L_ * D_ * DFF_ * 2);
    unsigned short* f2p  = (unsigned short*)alloc((size_t)L_ * DFF_ * D_ * 2);
    unsigned short* c1p  = (unsigned short*)alloc(384 * 64 * 2);
    unsigned short* c2p  = (unsigned short*)alloc(576 * 64 * 2);
    unsigned short* skp  = (unsigned short*)alloc(64 * 64 * 2);
    unsigned short* m1p  = (unsigned short*)alloc(576 * 64 * 2);
    unsigned short* m2p  = (unsigned short*)alloc(576 * 64 * 2);
    float* pfb = (float*)alloc((size_t)NP * D_ * 4);
    float* tok = (float*)alloc((size_t)NP * 32 * 4);
    unsigned short* Xin = (unsigned short*)alloc((size_t)NP * CIN_ * NPIX * 2);
    float* bufA = (float*)alloc((size_t)NP * CH_ * NPIX * 4);
    float* bufB = (float*)alloc((size_t)NP * CH_ * NPIX * 4);
    unsigned short* bufC = (unsigned short*)alloc((size_t)NP * CH_ * NPIX * 2);
    float* gmB = (float*)alloc((size_t)NP * 8 * 4);
    float* grB = (float*)alloc((size_t)NP * 8 * 4);

    // ---- pack weights to WMMA fragment layouts ----
    for (int l = 0; l < L_; ++l) {
        pack_wT_frag<<<(768 * 256 + 255) / 256, 256, 0, stream>>>(enc_wqkv + (size_t)l * 768 * 256, qkvp + (size_t)l * 256 * 768, 768, 256);
        pack_wT_frag<<<(256 * 256 + 255) / 256, 256, 0, stream>>>(enc_wo + (size_t)l * 256 * 256, wop + (size_t)l * 256 * 256, 256, 256);
        pack_wT_frag<<<(1024 * 256 + 255) / 256, 256, 0, stream>>>(fw1 + (size_t)l * 1024 * 256, f1p + (size_t)l * 256 * 1024, 1024, 256);
        pack_wT_frag<<<(256 * 1024 + 255) / 256, 256, 0, stream>>>(fw2 + (size_t)l * 256 * 1024, f2p + (size_t)l * 1024 * 256, 256, 1024);
    }
    pack_conv_frag<<<(384 * 64 + 255) / 256, 256, 0, stream>>>(c1w, c1p, 42, 3, 384);
    pack_conv_frag<<<(576 * 64 + 255) / 256, 256, 0, stream>>>(c2w, c2p, 64, 3, 576);
    pack_conv_frag<<<(64 * 64 + 255) / 256, 256, 0, stream>>>(skw, skp, 42, 1, 64);
    pack_conv_frag<<<(576 * 64 + 255) / 256, 256, 0, stream>>>(m1w, m1p, 64, 3, 576);
    pack_conv_frag<<<(576 * 64 + 255) / 256, 256, 0, stream>>>(m2w, m2p, 64, 3, 576);

    // ---- node features -> hidden ----
    node_embed<<<NT, 256, 0, stream>>>(node_features, cent, bbox, emb0, emb1, emb2,
                                       cont_w1, cont_b1, cont_w2, cont_b2, nh);
    message_k<<<NT, 256, 0, stream>>>(nh, nidx, ntok, nmask, edge_emb, pos_emb, hid);

    // ---- transformer layers (all GEMMs: fragment-packed A and B, no LDS) ----
    for (int l = 0; l < L_; ++l) {
        ln_kernel<<<NT, 256, 0, stream>>>(hid, ln1s + l * D_, ln1b + l * D_, nullptr, tAbf);
        gemm_bf16<<<dim3(4, 12), 256, 0, stream>>>(tAbf, qkvp + (size_t)l * 256 * 768, 768, 256,
                                                   enc_bqkv + l * 768, nullptr, tF1, nullptr, 768, 0);
        attn_kernel<<<B_ * H_, 128, 0, stream>>>(tF1, face_mask, tBbf);
        gemm_bf16<<<dim3(4, 4), 256, 0, stream>>>(tBbf, wop + (size_t)l * 256 * 256, 256, 256,
                                                  enc_bo + l * D_, hid, hid, nullptr, 256, 0);
        ln_kernel<<<NT, 256, 0, stream>>>(hid, ln2s + l * D_, ln2b + l * D_, nullptr, tAbf);
        gemm_bf16<<<dim3(4, 16), 256, 0, stream>>>(tAbf, f1p + (size_t)l * 256 * 1024, 1024, 256,
                                                   fb1 + l * DFF_, nullptr, nullptr, tBbf, 1024, 1);
        gemm_bf16<<<dim3(4, 4), 256, 0, stream>>>(tBbf, f2p + (size_t)l * 1024 * 256, 256, 1024,
                                                  fb2 + l * D_, hid, hid, nullptr, 256, 0);
    }
    ln_kernel<<<NT, 256, 0, stream>>>(hid, fn_s, fn_b, hidf, nullptr);
    hipMemcpyAsync(d_out, hidf, (size_t)NT * D_ * sizeof(float), hipMemcpyDeviceToDevice, stream);

    // ---- pair branch ----
    pair_pf_kernel<<<NP, 256, 0, stream>>>(pfeat, pf_w1, pf_b1, pf_w2, pf_b2, pfb);
    pair_mlp_kernel<<<NP, 256, 0, stream>>>(hidf, pfb, pidx, border_hidden,
                                            pm_w1, pm_b1, pm_w2, pm_b2, tp_w, tp_b, tok);
    build_maps_kernel<<<NP * CIN_, 256, 0, stream>>>(tok, pidx, pfeat, cent, bbox, Xin);

    // ---- CNN ----
    conv_wmma<3, 1, 42><<<NP * 128, 256, 0, stream>>>(Xin, c1p, c1b, bufA);
    gn_stats<<<NP * 8, 256, 0, stream>>>(bufA, gmB, grB);
    gn_apply<<<NP * 64, 256, 0, stream>>>(bufA, gmB, grB, g1s, g1b, nullptr, nullptr, bufC);
    conv_wmma<1, 0, 42><<<NP * 128, 256, 0, stream>>>(Xin, skp, skb, bufB);
    conv_wmma<3, 1, 64><<<NP * 128, 256, 0, stream>>>(bufC, c2p, c2b, bufA);
    gn_stats<<<NP * 8, 256, 0, stream>>>(bufA, gmB, grB);
    gn_apply<<<NP * 64, 256, 0, stream>>>(bufA, gmB, grB, g2s, g2b, bufB, bufB, bufC);
    conv_wmma<3, 1, 64><<<NP * 128, 256, 0, stream>>>(bufC, m1p, m1b, bufA);
    gn_stats<<<NP * 8, 256, 0, stream>>>(bufA, gmB, grB);
    gn_apply<<<NP * 64, 256, 0, stream>>>(bufA, gmB, grB, mg1s, mg1b, nullptr, nullptr, bufC);
    conv_wmma<3, 1, 64><<<NP * 128, 256, 0, stream>>>(bufC, m2p, m2b, bufA);
    gn_stats<<<NP * 8, 256, 0, stream>>>(bufA, gmB, grB);
    gn_apply<<<NP * 64, 256, 0, stream>>>(bufA, gmB, grB, mg2s, mg2b, bufB, bufA, nullptr);
    out_conv<<<(NP * NPIX + 255) / 256, 256, 0, stream>>>(bufA, ow, ob, pvalid,
                                                          (float*)d_out + (size_t)NT * D_);
}
